// MelBandMask_78872779424085
// MI455X (gfx1250) — compile-verified, matrix operand order checked by reading, not run on hardware
//
#include <hip/hip_runtime.h>

// MelBandMask fused kernel for gfx1250 (MI455X).
// Shapes: B=4, T=512 (M=2048 rows), K=63 bands, D=384, DFF=1536, OUT2=96.

#define KBANDS 63
#define Dd     384
#define DFFd   1536
#define OUT2   96
#define BW     48
#define MTOT   2048
#define MT     64        // rows per workgroup
#define NFREQ  1016
#define TT     512

typedef __attribute__((ext_vector_type(16))) _Float16 v16h;
typedef __attribute__((ext_vector_type(8)))  _Float16 v8h;
typedef __attribute__((ext_vector_type(8)))  float    v8f;
typedef __attribute__((ext_vector_type(4)))  int      i32x4;

#define A_STRIDE  392                       // halves per row (784B, 16B aligned)
#define H_STRIDE  1552                      // halves per row (3104B, 16B aligned)
#define C2_STRIDE 100                       // floats per row
#define A_OFF     0
#define H_OFF     (MT * A_STRIDE * 2)       // 50176 bytes
#define SMEM_BYTES (H_OFF + MT * H_STRIDE * 2)  // 248832 bytes (<320KB/WGP)

// ---- gfx1250 fast-math helpers (guarded; fall back to libm) --------------
__device__ __forceinline__ float fast_tanh(float v) {
#if __has_builtin(__builtin_amdgcn_tanhf)
    return __builtin_amdgcn_tanhf(v);          // v_tanh_f32
#elif __has_builtin(__builtin_amdgcn_tanh_f32)
    return __builtin_amdgcn_tanh_f32(v);
#else
    return tanhf(v);
#endif
}

__device__ __forceinline__ float fast_sigmoid(float g) {
#if __has_builtin(__builtin_amdgcn_exp2f) && __has_builtin(__builtin_amdgcn_rcpf)
    float e = __builtin_amdgcn_exp2f(-g * 1.44269504088896340736f); // v_exp_f32
    return __builtin_amdgcn_rcpf(1.0f + e);                        // v_rcp_f32
#else
    return 1.0f / (1.0f + expf(-g));
#endif
}

#if __has_builtin(__builtin_amdgcn_global_load_async_to_lds_b128) && \
    __has_builtin(__builtin_amdgcn_s_wait_asynccnt)
#define USE_ASYNC_LDS 1
typedef __attribute__((address_space(1))) i32x4 as1_i32x4;  // global (prints __device__)
typedef __attribute__((address_space(3))) i32x4 as3_i32x4;  // LDS    (prints __shared__)
#else
#define USE_ASYNC_LDS 0
#endif

__global__ void zero_out_kernel(float* __restrict__ out, int n) {
    int i = blockIdx.x * blockDim.x + threadIdx.x;
    if (i < n) out[i] = 0.0f;
}

__global__ __launch_bounds__(256, 1)
void melband_fused(const float* __restrict__ x,
                   const float* __restrict__ gamma,
                   const float* __restrict__ W0,
                   const float* __restrict__ b0,
                   const float* __restrict__ W1,
                   const float* __restrict__ b1,
                   float* __restrict__ out) {
    extern __shared__ char smem[];
    _Float16* Alds   = (_Float16*)(smem + A_OFF);
    _Float16* Hlds   = (_Float16*)(smem + H_OFF);
    float*    Xstage = (float*)(smem + H_OFF);   // x staging overlays H (phase 0 only)
    float*    C2     = (float*)(smem + A_OFF);   // C2 overlays A (phase 2 only)

    const int band = blockIdx.y;
    const int m0   = blockIdx.x * MT;
    const int tid  = threadIdx.x;
    const int lane = tid & 31;
    const int wave = tid >> 5;

    // ---- Phase 0: stage x tile (64 x 384 f32) into LDS -------------------
#if USE_ASYNC_LDS
    for (int i = tid; i < MT * (Dd / 4); i += 256) {
        int r  = i / (Dd / 4);
        int c4 = i % (Dd / 4);
        const float* g = x + ((size_t)(m0 + r) * KBANDS + band) * Dd + c4 * 4;
        float*       l = Xstage + r * Dd + c4 * 4;
        __builtin_amdgcn_global_load_async_to_lds_b128(
            (as1_i32x4*)g, (as3_i32x4*)l, 0, 0);
    }
    __builtin_amdgcn_s_wait_asynccnt(0);
#else
    for (int i = tid; i < MT * (Dd / 4); i += 256) {
        int r  = i / (Dd / 4);
        int c4 = i % (Dd / 4);
        float4 v = *(const float4*)(x + ((size_t)(m0 + r) * KBANDS + band) * Dd + c4 * 4);
        *(float4*)(Xstage + r * Dd + c4 * 4) = v;
    }
#endif
    __syncthreads();

    // ---- Phase 0b: RMSNorm * gamma -> A (f16) ----------------------------
    {
        int r = tid >> 2;      // row 0..63
        int q = tid & 3;       // quarter of 384
        const float* xr = Xstage + r * Dd + q * 96;
        float s = 0.0f;
        #pragma unroll 8
        for (int j = 0; j < 96; ++j) { float v = xr[j]; s += v * v; }
        s += __shfl_xor(s, 1, 32);
        s += __shfl_xor(s, 2, 32);
        float rs = rsqrtf(s * (1.0f / (float)Dd) + 1e-8f);
        const float* gk = gamma + band * Dd + q * 96;
        _Float16* ar = Alds + r * A_STRIDE + q * 96;
        #pragma unroll 8
        for (int j = 0; j < 96; ++j) ar[j] = (_Float16)(xr[j] * rs * gk[j]);
    }
    __syncthreads();

    const int cl   = lane & 15;
    const int koff = (lane < 16) ? 0 : 8;   // K-sub-offset per ISA 16-bit A/B layout

    // ---- Phase 1: GEMM1  C1 = A(64x384) * W0[k](384x1536), tanh -> H f16 -
    for (int nt = 0; nt < 12; ++nt) {
        const int n0   = wave * 192 + nt * 16;
        const int ncol = n0 + cl;
        v8f acc0 = {}, acc1 = {}, acc2 = {}, acc3 = {};
        const float* Wcol = W0 + (size_t)band * Dd * DFFd + ncol;
        for (int ks = 0; ks < 12; ++ks) {
            const int kk = ks * 32;
            v16h bm;
            #pragma unroll
            for (int i = 0; i < 8; ++i) {
                int rb = kk + koff + ((i < 4) ? 2 * i : 16 + 2 * (i - 4));
                bm[2 * i]     = (_Float16)Wcol[(size_t)rb * DFFd];
                bm[2 * i + 1] = (_Float16)Wcol[(size_t)(rb + 1) * DFFd];
            }
            #pragma unroll
            for (int mt = 0; mt < 4; ++mt) {
                const _Float16* ap = Alds + (mt * 16 + cl) * A_STRIDE + kk + koff;
                v8h lo = *(const v8h*)ap;
                v8h hi = *(const v8h*)(ap + 16);
                v16h am = __builtin_shufflevector(lo, hi,
                    0,1,2,3,4,5,6,7,8,9,10,11,12,13,14,15);
                v8f& acc = (mt == 0) ? acc0 : (mt == 1) ? acc1 : (mt == 2) ? acc2 : acc3;
                acc = __builtin_amdgcn_wmma_f32_16x16x32_f16(
                    false, am, false, bm, (short)0, acc, false, false);
            }
        }
        float bias = b0[band * DFFd + ncol];
        #pragma unroll
        for (int mt = 0; mt < 4; ++mt) {
            v8f acc = (mt == 0) ? acc0 : (mt == 1) ? acc1 : (mt == 2) ? acc2 : acc3;
            #pragma unroll
            for (int j = 0; j < 8; ++j) {
                int m = mt * 16 + j + ((lane < 16) ? 0 : 8);
                Hlds[m * H_STRIDE + ncol] = (_Float16)fast_tanh(acc[j] + bias);
            }
        }
    }
    __syncthreads();

    // ---- Phase 2: GEMM2  C2 = H(64x1536) * W1[k](1536x96) + b1 -> LDS f32
    for (int t3 = 0; t3 < 3; ++t3) {
        int t  = wave * 3 + t3;       // 24 tiles: 4 m-subtiles x 6 n-tiles
        int mt = t / 6;
        int nt = t % 6;
        int ncol = nt * 16 + cl;
        v8f acc = {};
        const float* Wcol = W1 + (size_t)band * DFFd * OUT2 + ncol;
        for (int ks = 0; ks < 48; ++ks) {
            const int kk = ks * 32;
            v16h bm;
            #pragma unroll
            for (int i = 0; i < 8; ++i) {
                int rb = kk + koff + ((i < 4) ? 2 * i : 16 + 2 * (i - 4));
                bm[2 * i]     = (_Float16)Wcol[(size_t)rb * OUT2];
                bm[2 * i + 1] = (_Float16)Wcol[(size_t)(rb + 1) * OUT2];
            }
            const _Float16* ap = Hlds + (mt * 16 + cl) * H_STRIDE + kk + koff;
            v8h lo = *(const v8h*)ap;
            v8h hi = *(const v8h*)(ap + 16);
            v16h am = __builtin_shufflevector(lo, hi,
                0,1,2,3,4,5,6,7,8,9,10,11,12,13,14,15);
            acc = __builtin_amdgcn_wmma_f32_16x16x32_f16(
                false, am, false, bm, (short)0, acc, false, false);
        }
        float bias = b1[band * OUT2 + ncol];
        #pragma unroll
        for (int j = 0; j < 8; ++j) {
            int m = mt * 16 + j + ((lane < 16) ? 0 : 8);
            C2[m * C2_STRIDE + ncol] = acc[j] + bias;
        }
    }
    __syncthreads();

    // ---- Phase 3: GLU + overlapped scatter-add ---------------------------
    for (int i = tid; i < MT * BW; i += 256) {
        int r = i / BW;
        int j = i % BW;
        float a = C2[r * C2_STRIDE + j];
        float g = C2[r * C2_STRIDE + j + BW];
        float val = a * fast_sigmoid(g);
        int m    = m0 + r;
        int bidx = m >> 9;          // m / 512
        int tt   = m & 511;         // m % 512
        int fc   = band * 32 + j;   // freq-channel column
        int f    = fc >> 1;
        int c    = fc & 1;
        atomicAdd(out + ((((size_t)bidx * 2 + c) * TT + tt) * NFREQ + f), val);
    }
}

extern "C" void kernel_launch(void* const* d_in, const int* in_sizes, int n_in,
                              void* d_out, int out_size, void* d_ws, size_t ws_size,
                              hipStream_t stream) {
    (void)in_sizes; (void)n_in; (void)d_ws; (void)ws_size;
    const float* x     = (const float*)d_in[0];
    const float* gamma = (const float*)d_in[1];
    const float* W0    = (const float*)d_in[2];
    const float* b0    = (const float*)d_in[3];
    const float* W1    = (const float*)d_in[4];
    const float* b1    = (const float*)d_in[5];
    float* out = (float*)d_out;

    zero_out_kernel<<<(out_size + 255) / 256, 256, 0, stream>>>(out, out_size);

    dim3 grid(MTOT / MT, KBANDS);   // x-fastest launch keeps same-band blocks
                                    // concurrent -> W0[k] served from L2
    melband_fused<<<grid, 256, SMEM_BYTES, stream>>>(x, gamma, W0, b0, W1, b1, out);
}
